// GAT__80942953660642
// MI455X (gfx1250) — compile-verified, matrix operand order checked by reading, not run on hardware
//
#include <hip/hip_runtime.h>

typedef __attribute__((ext_vector_type(16))) __bf16       v16bf;
typedef __attribute__((ext_vector_type(8)))  float        v8f;
typedef __attribute__((ext_vector_type(8)))  unsigned int v8u;
typedef __attribute__((ext_vector_type(4)))  unsigned int u32x4;
typedef __attribute__((ext_vector_type(8)))  int          i32x8;
typedef __attribute__((ext_vector_type(4)))  int          i32x4;

#define NEG_SLOPE 0.2f
#define SAF 34     // fp32 LDS row stride for A (32 + 2 pad DWORDs from TDM)
#define SB  36     // bf16 LDS row stride for B^T

static __device__ __forceinline__ unsigned short f2bf(float f) {
    unsigned u = __float_as_uint(f);
    u += 0x7FFFu + ((u >> 16) & 1u);          // round-to-nearest-even
    return (unsigned short)(u >> 16);
}

// monotonic float -> uint encoding so atomicMax(u32) == float max
static __device__ __forceinline__ unsigned enc_f32(float f) {
    unsigned u = __float_as_uint(f);
    return (u & 0x80000000u) ? ~u : (u | 0x80000000u);
}
static __device__ __forceinline__ float dec_f32(unsigned u) {
    return (u & 0x80000000u) ? __uint_as_float(u & 0x7FFFFFFFu)
                             : __uint_as_float(~u);
}

// ---------------------------------------------------------------------------
// TDM: DMA a 2D fp32 tile (tile_d0 elems x tile_d1 rows) from global to LDS,
// zero-filling past (tensor_d0, tensor_d1) remaining extents, inserting
// 2 DWORDs of LDS padding every 32 DWORDs (row stride 32 -> 34).
// Descriptor layout per CDNA5 ISA.
// ---------------------------------------------------------------------------
static __device__ __forceinline__ void tdm_load_2d_f32(
    unsigned lds_byte_off, const float* gaddr,
    unsigned tensor_d0, unsigned tensor_d1,
    unsigned tile_d0, unsigned tile_d1,
    unsigned long long stride0_elems)
{
    unsigned long long ga = (unsigned long long)(size_t)gaddr;

    u32x4 g0u;
    g0u[0] = 1u;                                            // count=1, user mode
    g0u[1] = lds_byte_off;                                  // lds_addr
    g0u[2] = (unsigned)(ga & 0xFFFFFFFFull);                // global_addr[31:0]
    g0u[3] = (unsigned)((ga >> 32) & 0x01FFFFFFull)         // global_addr[56:32]
           | (2u << 30);                                    // type=2 (image)

    unsigned w0 = (2u << 16)        // data_size = 4 bytes
                | (1u << 20)        // pad_enable
                | (4u << 22)        // pad_interval code 4 = every 32 DWORDs
                | (1u << 25);       // pad_amount  code 1 = 2 DWORDs
    unsigned w1 = (tensor_d0 & 0xFFFFu) << 16;              // dim0[15:0]
    unsigned w2 = ((tensor_d0 >> 16) & 0xFFFFu)             // dim0[31:16]
                | ((tensor_d1 & 0xFFFFu) << 16);            // dim1[15:0]
    unsigned w3 = ((tensor_d1 >> 16) & 0xFFFFu)             // dim1[31:16]
                | ((tile_d0 & 0xFFFFu) << 16);              // tile_dim0
    unsigned w4 = (tile_d1 & 0xFFFFu);                      // tile_dim1 (dim2=0)
    unsigned w5 = (unsigned)(stride0_elems & 0xFFFFFFFFull);
    unsigned w6 = (unsigned)((stride0_elems >> 32) & 0xFFFFull);
    unsigned w7 = 0u;

    i32x8 g1;
    g1[0] = (int)w0; g1[1] = (int)w1; g1[2] = (int)w2; g1[3] = (int)w3;
    g1[4] = (int)w4; g1[5] = (int)w5; g1[6] = (int)w6; g1[7] = (int)w7;

    i32x4 gz = {};
#if defined(__clang_major__) && (__clang_major__ >= 23)
    i32x8 gz8 = {};
    __builtin_amdgcn_tensor_load_to_lds(g0u, g1, gz, gz, gz8, 0);
#else
    __builtin_amdgcn_tensor_load_to_lds(g0u, g1, gz, gz, 0);
#endif
}

// ---------------------------------------------------------------------------
// GEMM: C[N,M] = A[N,K] * B[K,M], fp32 in / fp32 out, bf16 WMMA accumulate.
// Block: 256 threads = 8 waves; block covers 128 rows x (NT*16) cols.
// A tiles DMA'd by TDM into double-buffered LDS (fp32, stride SAF), one chunk
// ahead of compute; B tiles staged manually (transposed, bf16, double buffer).
// One workgroup barrier per K-chunk.
// ---------------------------------------------------------------------------
template<int NT>
__global__ __launch_bounds__(256)
void gemm_bf16_wmma(const float* __restrict__ A, const float* __restrict__ B,
                    float* __restrict__ C, int N, int K, int M)
{
    __shared__ float          sAf[2][128 * SAF + 4];     // fp32 A tiles (TDM dest)
    __shared__ unsigned short sBT[2][NT * 16 * SB];      // bf16 B^T tiles

    const int tid  = threadIdx.x;
    const int lane = tid & 31;
    const int wave = tid >> 5;
    const int row0 = blockIdx.x * 128;

    v8f acc[NT];
    const v8f vzero = {};
#pragma unroll
    for (int t = 0; t < NT; ++t) acc[t] = vzero;

    const int mloc = wave * 16 + (lane & 15);        // row within block
    const int kb   = (lane < 16) ? 0 : 8;            // per-lane K base (ISA layout)
    const int nloc = lane & 15;

    const int nchunks = (K + 31) >> 5;

    const unsigned rows_rem = (unsigned)((N > row0) ? (N - row0) : 0);

    // stage B chunk `ch` into buffer `buf` (transposed, bf16, zero-padded)
    auto stageB = [&](int ch, int buf) {
        const int k0 = ch << 5;
        for (int idx = tid; idx < 32 * NT * 16; idx += 256) {
            int k = idx / (NT * 16), c = idx % (NT * 16);
            int gk = k0 + k;
            float v = (gk < K && c < M) ? B[(long long)gk * M + c] : 0.0f;
            sBT[buf][c * SB + k] = f2bf(v);
        }
    };

    // prologue: DMA chunk 0 of A, stage chunk 0 of B
    if (wave == 0) {
        tdm_load_2d_f32((unsigned)(size_t)&sAf[0][0],
                        A + (long long)row0 * K,
                        (unsigned)K, rows_rem, 32u, 128u,
                        (unsigned long long)K);
    }
    stageB(0, 0);
    if (wave == 0) __builtin_amdgcn_s_wait_tensorcnt(0);
    __syncthreads();

    for (int ch = 0; ch < nchunks; ++ch) {
        const int buf = ch & 1;

        // prefetch chunk ch+1 while computing chunk ch
        if (ch + 1 < nchunks) {
            const int nk0 = (ch + 1) << 5;
            if (wave == 0) {
                tdm_load_2d_f32((unsigned)(size_t)&sAf[buf ^ 1][0],
                                A + (long long)row0 * K + nk0,
                                (unsigned)(K - nk0), rows_rem, 32u, 128u,
                                (unsigned long long)K);
            }
            stageB(ch + 1, buf ^ 1);
        }

        // A fragment: 16-bit A 16x32 layout (VGPR j holds K = kb+2j (+8 if j>=4)),
        // converting fp32 LDS data to packed bf16 on the fly
        v8u au;
#pragma unroll
        for (int j = 0; j < 8; ++j) {
            int kj = kb + 2 * j + ((j >= 4) ? 8 : 0);
            float2 f = *(const float2*)&sAf[buf][mloc * SAF + kj];
            au[j] = (unsigned)f2bf(f.x) | ((unsigned)f2bf(f.y) << 16);
        }
        v16bf af = __builtin_bit_cast(v16bf, au);

#pragma unroll
        for (int t = 0; t < NT; ++t) {
            v8u bu;
#pragma unroll
            for (int j = 0; j < 8; ++j) {
                int kj = kb + 2 * j + ((j >= 4) ? 8 : 0);
                bu[j] = *(const unsigned int*)&sBT[buf][(t * 16 + nloc) * SB + kj];
            }
            v16bf bfr = __builtin_bit_cast(v16bf, bu);
            acc[t] = __builtin_amdgcn_wmma_f32_16x16x32_bf16(
                         false, af, false, bfr, (short)0, acc[t], false, false);
        }

        if (wave == 0) __builtin_amdgcn_s_wait_tensorcnt(0);
        __syncthreads();
    }

    // C/D layout: VGPR v -> M = v (lanes 0-15) / v+8 (lanes 16-31), N = lane&15
#pragma unroll
    for (int t = 0; t < NT; ++t) {
#pragma unroll
        for (int v = 0; v < 8; ++v) {
            int m  = (lane < 16) ? v : (v + 8);
            int gr = row0 + wave * 16 + m;
            int gc = t * 16 + nloc;
            if (gr < N && gc < M) C[(long long)gr * M + gc] = acc[t][v];
        }
    }
}

// ---------------------------------------------------------------------------
// per-node attention terms: el = h . al, er = h . ar  (one wave per node)
// ---------------------------------------------------------------------------
__global__ void node_attn(const float* __restrict__ h,
                          const float* __restrict__ al, const float* __restrict__ ar,
                          float* __restrict__ el, float* __restrict__ er,
                          int N, int d)
{
    int node = blockIdx.x * 8 + (threadIdx.x >> 5);
    int lane = threadIdx.x & 31;
    if (node >= N) return;
    float pl = 0.f, pr = 0.f;
    for (int c = lane; c < d; c += 32) {
        float v = h[(long long)node * d + c];
        pl += v * al[c];
        pr += v * ar[c];
    }
    for (int off = 16; off > 0; off >>= 1) {
        pl += __shfl_down(pl, off);
        pr += __shfl_down(pr, off);
    }
    if (lane == 0) { el[node] = pl; er[node] = pr; }
}

__global__ void fill_f32(float* p, float v, int n) {
    int i = blockIdx.x * blockDim.x + threadIdx.x;
    if (i < n) p[i] = v;
}
__global__ void fill_u32(unsigned* p, unsigned v, int n) {
    int i = blockIdx.x * blockDim.x + threadIdx.x;
    if (i < n) p[i] = v;
}

static __device__ __forceinline__ float edge_e(const float* el, const float* er,
                                               int s, int t) {
    float x = el[s] + er[t];
    return (x > 0.f) ? x : x * NEG_SLOPE;
}

__global__ void edge_max(const int* __restrict__ src, const int* __restrict__ dst,
                         const float* __restrict__ el, const float* __restrict__ er,
                         unsigned* __restrict__ mu, int E)
{
    int e = blockIdx.x * blockDim.x + threadIdx.x;
    if (e >= E) return;
    int s = src[e], t = dst[e];
    atomicMax(&mu[t], enc_f32(edge_e(el, er, s, t)));
}

__global__ void edge_exp(const int* __restrict__ src, const int* __restrict__ dst,
                         const float* __restrict__ el, const float* __restrict__ er,
                         const unsigned* __restrict__ mu,
                         float* __restrict__ ex, float* __restrict__ denom, int E)
{
    int e = blockIdx.x * blockDim.x + threadIdx.x;
    if (e >= E) return;
    int s = src[e], t = dst[e];
    float v = __expf(edge_e(el, er, s, t) - dec_f32(mu[t]));
    ex[e] = v;
    unsafeAtomicAdd(&denom[t], v);
}

// scatter: out[dst] += h[src] * (ex/denom[dst]); one thread per (edge, channel)
__global__ void edge_scatter(const int* __restrict__ src, const int* __restrict__ dst,
                             const float* __restrict__ h, const float* __restrict__ ex,
                             const float* __restrict__ denom, float* __restrict__ out,
                             int E, int d)
{
    long long idx = (long long)blockIdx.x * blockDim.x + threadIdx.x;
    long long tot = (long long)E * d;
    if (idx >= tot) return;
    int e = (int)(idx / d);
    int c = (int)(idx - (long long)e * d);
    int s = src[e], t = dst[e];
    float w = ex[e] / denom[t];
    unsafeAtomicAdd(&out[(long long)t * d + c], h[(long long)s * d + c] * w);
}

__global__ void bias_relu(float* __restrict__ io, const float* __restrict__ b,
                          int N, int d)
{
    long long idx = (long long)blockIdx.x * blockDim.x + threadIdx.x;
    if (idx >= (long long)N * d) return;
    float v = io[idx] + b[(int)(idx % d)];
    io[idx] = (v > 0.f) ? v : 0.f;
}

// ---------------------------------------------------------------------------
extern "C" void kernel_launch(void* const* d_in, const int* in_sizes, int n_in,
                              void* d_out, int out_size, void* d_ws, size_t ws_size,
                              hipStream_t stream)
{
    const float* X   = (const float*)d_in[0];
    const int*   src = (const int*)d_in[1];
    const int*   dst = (const int*)d_in[2];
    const float* W0  = (const float*)d_in[3];
    const float* al0 = (const float*)d_in[4];
    const float* ar0 = (const float*)d_in[5];
    const float* b0  = (const float*)d_in[6];
    const float* W1  = (const float*)d_in[7];
    const float* al1 = (const float*)d_in[8];
    const float* ar1 = (const float*)d_in[9];
    const float* b1  = (const float*)d_in[10];

    const int D0 = in_sizes[4];           // 140
    const int K0 = in_sizes[3] / D0;      // 1433
    const int N  = in_sizes[0] / K0;      // 100000
    const int E  = in_sizes[1];           // 1,700,000
    const int D1 = in_sizes[8];           // 7

    float* ws     = (float*)d_ws;
    float* H      = ws;                               // N*D0 (reused as N*D1)
    float* AGG    = H + (size_t)N * D0;               // N*D0
    float* el     = AGG + (size_t)N * D0;             // N
    float* er     = el + N;                           // N
    unsigned* mu  = (unsigned*)(er + N);              // N
    float* denom  = (float*)mu + N;                   // N
    float* ex     = denom + N;                        // E
    float* outF   = (float*)d_out;

    const unsigned ENC_NEG_INF = 0x007FFFFFu;         // enc(-inf)
    dim3 blk(256);
    int gridRows = (N + 127) / 128;

    // ---------------- Layer 0 (K=1433 -> D0=140) ----------------
    gemm_bf16_wmma<9><<<gridRows, blk, 0, stream>>>(X, W0, H, N, K0, D0);
    node_attn<<<(N + 7) / 8, blk, 0, stream>>>(H, al0, ar0, el, er, N, D0);
    fill_u32<<<(N + 255) / 256, blk, 0, stream>>>(mu, ENC_NEG_INF, N);
    fill_f32<<<(N + 255) / 256, blk, 0, stream>>>(denom, 0.f, N);
    fill_f32<<<((size_t)N * D0 + 255) / 256, blk, 0, stream>>>(AGG, 0.f, N * D0);
    edge_max<<<(E + 255) / 256, blk, 0, stream>>>(src, dst, el, er, mu, E);
    edge_exp<<<(E + 255) / 256, blk, 0, stream>>>(src, dst, el, er, mu, ex, denom, E);
    edge_scatter<<<(unsigned)(((long long)E * D0 + 255) / 256), blk, 0, stream>>>(
        src, dst, H, ex, denom, AGG, E, D0);
    bias_relu<<<(unsigned)(((long long)N * D0 + 255) / 256), blk, 0, stream>>>(
        AGG, b0, N, D0);

    // ---------------- Layer 1 (K=140 -> D1=7) ----------------
    gemm_bf16_wmma<1><<<gridRows, blk, 0, stream>>>(AGG, W1, H, N, D0, D1);
    node_attn<<<(N + 7) / 8, blk, 0, stream>>>(H, al1, ar1, el, er, N, D1);
    fill_u32<<<(N + 255) / 256, blk, 0, stream>>>(mu, ENC_NEG_INF, N);
    fill_f32<<<(N + 255) / 256, blk, 0, stream>>>(denom, 0.f, N);
    fill_f32<<<((size_t)N * D1 + 255) / 256, blk, 0, stream>>>(outF, 0.f, N * D1);
    edge_max<<<(E + 255) / 256, blk, 0, stream>>>(src, dst, el, er, mu, E);
    edge_exp<<<(E + 255) / 256, blk, 0, stream>>>(src, dst, el, er, mu, ex, denom, E);
    edge_scatter<<<(unsigned)(((long long)E * D1 + 255) / 256), blk, 0, stream>>>(
        src, dst, H, ex, denom, outF, E, D1);
    bias_relu<<<(unsigned)(((long long)N * D1 + 255) / 256), blk, 0, stream>>>(
        outF, b1, N, D1);
}
